// GraphEmbeddings_24103356465597
// MI455X (gfx1250) — matrix operand order repaired and checked by reading, not compiled, not run on hardware
//
#include <hip/hip_runtime.h>
#include <hip/hip_bf16.h>

// ---------------------------------------------------------------------------
// CGCNN forward for MI455X (gfx1250, wave32, WMMA).
// Conv GEMM (1.2M x 169 x 128) done with v_wmma_f32_16x16x32_bf16, two-pass
// batchnorm with recompute (compute is free vs 23.3 TB/s HBM), FC computed
// only on gathered rows. Workspace: ~51.3 MB.
// ---------------------------------------------------------------------------

typedef __attribute__((ext_vector_type(16))) __bf16 v16bf;
typedef __attribute__((ext_vector_type(8)))  float  v8f;

#define N_ATOMS 100000
#define MNBR    12
#define FDIM    64
#define NBRF    41
#define KTOT    169      // 2*64 + 41
#define C2      128      // 2*FDIM
#define HID     768
#define BATCH   32
#define GLEN    512
#define NROWS   (N_ATOMS * MNBR)   // 1,200,000 = 9375 * 128 exactly
#define EPSBN   1e-5f

__device__ __forceinline__ float sigmoidf_(float x) {
    return 1.0f / (1.0f + __expf(-x));
}
__device__ __forceinline__ float softplusf_(float x) {
    return (x > 20.0f) ? x : __logf(1.0f + __expf(x));
}

// element (row, gk) of the virtual "total" matrix [self | nbr | edge | 0pad]
__device__ __forceinline__ float total_elem(const float* __restrict__ fea,
                                            const int* __restrict__ nbr_idx,
                                            const float* __restrict__ nbr_fea,
                                            int n, int m, int gk) {
    if (gk < FDIM)        return fea[n * FDIM + gk];
    if (gk < 2 * FDIM)    return fea[nbr_idx[n * MNBR + m] * FDIM + (gk - FDIM)];
    if (gk < KTOT)        return nbr_fea[(n * MNBR + m) * NBRF + (gk - 2 * FDIM)];
    return 0.0f;
}

// ------------------------------- embedding ---------------------------------
__global__ __launch_bounds__(256) void k_embed(const int* __restrict__ atom_num,
                                               const float* __restrict__ emb,
                                               float* __restrict__ fea) {
    for (long long i = (long long)blockIdx.x * 256 + threadIdx.x;
         i < (long long)N_ATOMS * FDIM; i += (long long)gridDim.x * 256) {
        int n = (int)(i >> 6), f = (int)(i & 63);
        fea[i] = emb[atom_num[n] * FDIM + f];
    }
}

// --------------------------- conv GEMM (WMMA) ------------------------------
// Block: 256 thr (8 waves), tile = 128 rows x 128 cols, K padded to 192.
// Wave w owns rows [w*16, w*16+16) and all 8 column tiles.
// STATS_ONLY: accumulate per-channel sum/sumsq of gated (for BN1).
// else: apply BN1, sigmoid(filter)*softplus(core), atomic-sum over M.
template <bool STATS_ONLY>
__global__ __launch_bounds__(256) void k_conv_gemm(
    const float* __restrict__ fea, const int* __restrict__ nbr_idx,
    const float* __restrict__ nbr_fea, const float* __restrict__ W,
    const float* __restrict__ bias, float* __restrict__ stats1,
    const float* __restrict__ bnp1, float* __restrict__ summed) {
    __shared__ __bf16 lA[128][66];   // [row][k]   (padded)
    __shared__ __bf16 lB[128][66];   // [col][k]   (W transposed)
    __shared__ float ssum[C2], ssq[C2];

    const int tid = threadIdx.x;
    const int wv = tid >> 5, lane = tid & 31;
    const int l2 = lane & 15, hi = lane >> 4;
    const int rowBase = blockIdx.x * 128;

    if (STATS_ONLY && tid < C2) { ssum[tid] = 0.0f; ssq[tid] = 0.0f; }

    v8f acc[8] = {};

    for (int kc = 0; kc < 3; ++kc) {
        const int k0 = kc * 64;
        __syncthreads();
        // stage A chunk: 128 rows x 64 k (gather + f32->bf16)
        for (int idx = tid; idx < 128 * 64; idx += 256) {
            int r = idx >> 6, k = idx & 63;
            int row = rowBase + r;
            int n = row / MNBR, m = row - n * MNBR;
            lA[r][k] = (__bf16)total_elem(fea, nbr_idx, nbr_fea, n, m, k0 + k);
        }
        // stage B chunk transposed: [col][k] = W[(k0+k)*128 + col]
        for (int idx = tid; idx < 128 * 64; idx += 256) {
            int c = idx >> 6, k = idx & 63;
            int gk = k0 + k;
            float v = (gk < KTOT) ? W[gk * C2 + c] : 0.0f;
            lB[c][k] = (__bf16)v;
        }
        if (kc < 2) __builtin_prefetch(&W[(k0 + 64) * C2], 0, 1);
        __syncthreads();

#pragma unroll
        for (int ks = 0; ks < 64; ks += 32) {
            // A fragment: lane<16 holds K {0..7,16..23}, lane>=16 {8..15,24..31}
            union { v16bf v; unsigned int u[8]; } af;
#pragma unroll
            for (int p = 0; p < 8; ++p) {
                int kb = (p < 4 ? 2 * p : 16 + 2 * (p - 4)) + hi * 8;
                af.u[p] = *(const unsigned int*)&lA[wv * 16 + l2][ks + kb];
            }
#pragma unroll
            for (int tn = 0; tn < 8; ++tn) {
                union { v16bf v; unsigned int u[8]; } bfr;
#pragma unroll
                for (int p = 0; p < 8; ++p) {
                    int kb = (p < 4 ? 2 * p : 16 + 2 * (p - 4)) + hi * 8;
                    bfr.u[p] = *(const unsigned int*)&lB[tn * 16 + l2][ks + kb];
                }
                acc[tn] = __builtin_amdgcn_wmma_f32_16x16x32_bf16(
                    false, af.v, false, bfr.v, (short)0, acc[tn], false, false);
            }
        }
    }
    __syncthreads();

    if (STATS_ONLY) {
#pragma unroll
        for (int tn = 0; tn < 8; ++tn) {
            int ch = tn * 16 + l2;
            float bch = bias[ch];
            float bsum = 0.0f, bsq = 0.0f;
#pragma unroll
            for (int r = 0; r < 8; ++r) {
                float v = acc[tn][r] + bch;
                bsum += v;
                bsq += v * v;
            }
            atomicAdd(&ssum[ch], bsum);
            atomicAdd(&ssq[ch], bsq);
        }
        __syncthreads();
        if (tid < C2) {
            atomicAdd(&stats1[tid], ssum[tid]);
            atomicAdd(&stats1[C2 + tid], ssq[tid]);
        }
    } else {
#pragma unroll
        for (int tn = 0; tn < 4; ++tn) {
            int chF = tn * 16 + l2, chC = chF + FDIM;
            float bF = bias[chF], bC = bias[chC];
            float sF = bnp1[chF], hF = bnp1[C2 + chF];
            float sC = bnp1[chC], hC = bnp1[C2 + chC];
#pragma unroll
            for (int r = 0; r < 8; ++r) {
                int grow = rowBase + wv * 16 + (r + 8 * hi);
                int natom = grow / MNBR;
                float fb = (acc[tn][r] + bF) * sF + hF;       // BN1(filter)
                float cb = (acc[tn + 4][r] + bC) * sC + hC;   // BN1(core)
                float act = sigmoidf_(fb) * softplusf_(cb);
                atomicAdd(&summed[natom * FDIM + chF], act);  // sum over M
            }
        }
    }
}

// ---------------- finalize BN: stats -> (scale, shift) ---------------------
__global__ void k_fin_bn(const float* __restrict__ stats, const float* __restrict__ g,
                         const float* __restrict__ b, float* __restrict__ bnp,
                         int nch, float invcnt) {
    int c = threadIdx.x;
    if (c < nch) {
        float mean = stats[c] * invcnt;
        float var = stats[nch + c] * invcnt - mean * mean;
        float scale = rsqrtf(var + EPSBN) * g[c];
        bnp[c] = scale;
        bnp[nch + c] = b[c] - mean * scale;
    }
}

// ---------------- per-channel stats of summed[N,64] ------------------------
__global__ __launch_bounds__(256) void k_sum_stats(const float* __restrict__ summed,
                                                   float* __restrict__ stats2) {
    __shared__ float ssum[FDIM], ssq[FDIM];
    int tid = threadIdx.x;
    if (tid < FDIM) { ssum[tid] = 0.0f; ssq[tid] = 0.0f; }
    __syncthreads();
    for (long long i = (long long)blockIdx.x * 256 + tid;
         i < (long long)N_ATOMS * FDIM; i += (long long)gridDim.x * 256) {
        float v = summed[i];
        int ch = (int)(i & (FDIM - 1));
        atomicAdd(&ssum[ch], v);
        atomicAdd(&ssq[ch], v * v);
    }
    __syncthreads();
    if (tid < FDIM) {
        atomicAdd(&stats2[tid], ssum[tid]);
        atomicAdd(&stats2[FDIM + tid], ssq[tid]);
    }
}

// ---------------- residual: fea = softplus(fea + BN2(summed)) --------------
__global__ __launch_bounds__(256) void k_apply(float* __restrict__ fea,
                                               const float* __restrict__ summed,
                                               const float* __restrict__ bnp2) {
    for (long long i = (long long)blockIdx.x * 256 + threadIdx.x;
         i < (long long)N_ATOMS * FDIM; i += (long long)gridDim.x * 256) {
        int ch = (int)(i & (FDIM - 1));
        float s = summed[i] * bnp2[ch] + bnp2[FDIM + ch];
        fea[i] = softplusf_(fea[i] + s);
    }
}

// ---------------- FC on gathered rows only (16384 x 64 x 768) --------------
// Block: 8 waves, 16 rows (one A tile shared), each wave does 6 col tiles.
__global__ __launch_bounds__(256) void k_fc(
    const float* __restrict__ fea, const int* __restrict__ sel_idx,
    const float* __restrict__ mask, const float* __restrict__ fcW,
    const float* __restrict__ fcb, float* __restrict__ out) {
    __shared__ __bf16 lA[16][66];
    const int tid = threadIdx.x, wv = tid >> 5, lane = tid & 31;
    const int l2 = lane & 15, hi = lane >> 4;
    const int rowBase = blockIdx.x * 16;

    for (int idx = tid; idx < 16 * FDIM; idx += 256) {
        int r = idx >> 6, k = idx & 63;
        int atom = sel_idx[rowBase + r];
        lA[r][k] = (__bf16)fea[atom * FDIM + k];
    }
    __syncthreads();

    union { v16bf v; unsigned int u[8]; } af[2];
#pragma unroll
    for (int ks2 = 0; ks2 < 2; ++ks2)
#pragma unroll
        for (int p = 0; p < 8; ++p) {
            int kb = (p < 4 ? 2 * p : 16 + 2 * (p - 4)) + hi * 8;
            af[ks2].u[p] = *(const unsigned int*)&lA[l2][ks2 * 32 + kb];
        }

    v8f acc[6] = {};
#pragma unroll
    for (int tl = 0; tl < 6; ++tl) {
        int ct = wv * 6 + tl;
#pragma unroll
        for (int ks2 = 0; ks2 < 2; ++ks2) {
            v16bf bvec;
#pragma unroll
            for (int j = 0; j < 16; ++j) {
                int gk = ks2 * 32 + (j < 8 ? j : 8 + j) + hi * 8;
                bvec[j] = (__bf16)fcW[gk * HID + ct * 16 + l2];
            }
            acc[tl] = __builtin_amdgcn_wmma_f32_16x16x32_bf16(
                false, af[ks2].v, false, bvec, (short)0, acc[tl], false, false);
        }
    }

#pragma unroll
    for (int tl = 0; tl < 6; ++tl) {
        int col = (wv * 6 + tl) * 16 + l2;
        float bb = fcb[col];
#pragma unroll
        for (int r = 0; r < 8; ++r) {
            int grow = rowBase + r + 8 * hi;
            float v = (acc[tl][r] + bb) * mask[grow];
            out[(long long)grow * HID + col] = v;
        }
    }
}

// ---------------- tuple output #2: copy mask -------------------------------
__global__ void k_mask(const float* __restrict__ mask, float* __restrict__ out) {
    int i = blockIdx.x * 256 + threadIdx.x;
    if (i < BATCH * GLEN) out[i] = mask[i];
}

// ---------------------------------------------------------------------------
extern "C" void kernel_launch(void* const* d_in, const int* in_sizes, int n_in,
                              void* d_out, int out_size, void* d_ws, size_t ws_size,
                              hipStream_t stream) {
    (void)in_sizes; (void)n_in; (void)out_size; (void)ws_size;
    const int*   atom_num = (const int*)d_in[0];
    const int*   nbr_idx  = (const int*)d_in[1];
    const float* nbr_fea  = (const float*)d_in[2];
    const int*   sel_idx  = (const int*)d_in[3];
    const float* mask     = (const float*)d_in[4];
    const float* emb      = (const float*)d_in[5];
    const float* conv_W   = (const float*)d_in[6];
    const float* conv_b   = (const float*)d_in[7];
    const float* bn1_g    = (const float*)d_in[8];
    const float* bn1_b    = (const float*)d_in[9];
    const float* bn2_g    = (const float*)d_in[10];
    const float* bn2_b    = (const float*)d_in[11];
    const float* fc_W     = (const float*)d_in[12];
    const float* fc_b     = (const float*)d_in[13];
    float* out = (float*)d_out;

    // workspace carve-up (~51.3 MB)
    float* fea    = (float*)d_ws;                       // [N,64]
    float* summed = fea + (size_t)N_ATOMS * FDIM;       // [N,64]
    float* stats1 = summed + (size_t)N_ATOMS * FDIM;    // [256]
    float* stats2 = stats1 + 2 * C2;                    // [128]
    float* bnp1   = stats2 + 2 * FDIM;                  // [256]
    float* bnp2   = bnp1 + 2 * C2;                      // [128]

    k_embed<<<4096, 256, 0, stream>>>(atom_num, emb, fea);

    const int gemmGrid = NROWS / 128;  // 9375 exactly
    for (int i = 0; i < 3; ++i) {
        const float* W  = conv_W + (size_t)i * KTOT * C2;
        const float* bb = conv_b + (size_t)i * C2;
        hipMemsetAsync(stats1, 0, 2 * C2 * sizeof(float), stream);
        hipMemsetAsync(stats2, 0, 2 * FDIM * sizeof(float), stream);
        hipMemsetAsync(summed, 0, (size_t)N_ATOMS * FDIM * sizeof(float), stream);

        k_conv_gemm<true><<<gemmGrid, 256, 0, stream>>>(
            fea, nbr_idx, nbr_fea, W, bb, stats1, bnp1, summed);
        k_fin_bn<<<1, C2, 0, stream>>>(stats1, bn1_g + i * C2, bn1_b + i * C2,
                                       bnp1, C2, 1.0f / (float)NROWS);
        k_conv_gemm<false><<<gemmGrid, 256, 0, stream>>>(
            fea, nbr_idx, nbr_fea, W, bb, stats1, bnp1, summed);
        k_sum_stats<<<512, 256, 0, stream>>>(summed, stats2);
        k_fin_bn<<<1, FDIM, 0, stream>>>(stats2, bn2_g + i * FDIM, bn2_b + i * FDIM,
                                         bnp2, FDIM, 1.0f / (float)N_ATOMS);
        k_apply<<<4096, 256, 0, stream>>>(fea, summed, bnp2);
    }

    k_fc<<<(BATCH * GLEN) / 16, 256, 0, stream>>>(fea, sel_idx, mask, fc_W, fc_b, out);
    k_mask<<<(BATCH * GLEN + 255) / 256, 256, 0, stream>>>(
        mask, out + (size_t)BATCH * GLEN * HID);
}
